// MoleculeModel_88175678587214
// MI455X (gfx1250) — matrix operand order repaired and checked by reading, not compiled, not run on hardware
//
#include <hip/hip_runtime.h>

// ---------------------------------------------------------------------------
// MoleculeModel on MI455X (gfx1250, wave32, WMMA + double-buffered TDM)
//
//   1) hidden: fp32 -> bf16;  weights: fp32 -> bf16 *transposed* (W^T[Hout,K])
//   2) gemm_bias_relu_bf16: both A and B(=W^T) tiles staged by
//      TENSOR_LOAD_TO_LDS (pad fields reproduce the padded LDS layout),
//      double-buffered via TENSORcnt; 8x v_wmma_f32_16x16x32_bf16 per wave/step
//   3) gemv_rows (OUT == 1), then segment sums + final combine.
// ---------------------------------------------------------------------------

typedef __attribute__((ext_vector_type(16))) __bf16 v16bf;
typedef __attribute__((ext_vector_type(8)))  __bf16 v8bf;
typedef __attribute__((ext_vector_type(8)))  float  v8f;
typedef __attribute__((ext_vector_type(4)))  unsigned int v4u;
typedef __attribute__((ext_vector_type(8)))  int    v8i_;
typedef __attribute__((ext_vector_type(4)))  int    v4i_;

struct v8bf_pair { v8bf lo, hi; };

static __device__ __forceinline__ v16bf join16(v8bf lo, v8bf hi) {
    v8bf_pair p{lo, hi};
    return __builtin_bit_cast(v16bf, p);
}

// ---------------------------------------------------------------------------
// TDM helpers: 2D bf16 tile load, tile = 32 (contig) x 128 rows, with LDS
// padding of 4 DWORDs every 16 DWORDs -> row stride 40 bf16 (= LDT).
// ---------------------------------------------------------------------------
static __device__ __forceinline__ v8i_ tdm_desc_g1(int dim0, int dim1) {
    v8i_ g;
    g[0] = (1 << 16)    // data_size = 2 bytes
         | (1 << 20)    // pad_enable
         | (3 << 22)    // pad_interval: every 16 DWORDs (= 32 bf16 = tile_dim0)
         | (3 << 25);   // pad_amount: 4 DWORDs (= 8 bf16)
    g[1] = (int)((unsigned)dim0 << 16);                     // tensor_dim0[15:0]
    g[2] = (int)(((unsigned)dim0 >> 16) | ((unsigned)dim1 << 16));
    g[3] = (int)(((unsigned)dim1 >> 16) | (32u << 16));     // tile_dim0 = 32
    g[4] = 128;                                             // tile_dim1 = 128
    g[5] = dim0;                                            // dim0_stride = dim0
    g[6] = 0;
    g[7] = 0;
    return g;
}

static __device__ __forceinline__ void tdm_load_2d(unsigned int lds_addr,
                                                   unsigned long long ga,
                                                   v8i_ g1) {
    v4i_ gz = {0, 0, 0, 0};
    v4u g0;
    g0[0] = 1u;                                             // count=1, 2D
    g0[1] = lds_addr;
    g0[2] = (unsigned int)ga;
    g0[3] = (unsigned int)((ga >> 32) & 0x1FFFFFFu) | (2u << 30);  // type=2
#if __has_include(<hip/amd_detail/amd_gfx1250_TDM.h>)
    v8i_ g5 = {0, 0, 0, 0, 0, 0, 0, 0};
    __builtin_amdgcn_tensor_load_to_lds(g0, g1, gz, gz, g5, 0);
#else
    __builtin_amdgcn_tensor_load_to_lds(g0, g1, gz, gz, 0);
#endif
}

// ---------------------------------------------------------------------------
// fp32 -> bf16 conversion (grid-stride)
// ---------------------------------------------------------------------------
__global__ void cvt_f32_bf16(const float* __restrict__ src,
                             __bf16* __restrict__ dst, size_t n) {
    size_t i = (size_t)blockIdx.x * blockDim.x + threadIdx.x;
    size_t stride = (size_t)gridDim.x * blockDim.x;
    for (; i < n; i += stride) dst[i] = (__bf16)src[i];
}

// ---------------------------------------------------------------------------
// fp32[R,Ccols] -> bf16 transposed [Ccols,R]  (32x32 LDS-tiled, 256 threads)
// ---------------------------------------------------------------------------
__global__ __launch_bounds__(256)
void cvt_transpose_f32_bf16(const float* __restrict__ src,
                            __bf16* __restrict__ dst, int R, int Ccols) {
    __shared__ float tile[32][33];
    const int bx = blockIdx.x * 32;   // column base
    const int by = blockIdx.y * 32;   // row base
    const int tx = threadIdx.x & 31;
    const int ty = threadIdx.x >> 5;  // 0..7
#pragma unroll
    for (int r = ty; r < 32; r += 8)
        tile[r][tx] = src[(size_t)(by + r) * Ccols + bx + tx];
    __syncthreads();
#pragma unroll
    for (int r = ty; r < 32; r += 8)
        dst[(size_t)(bx + r) * R + by + tx] = (__bf16)tile[tx][r];
}

// ---------------------------------------------------------------------------
// C_bf16[Nrows,Hout] = act(A_bf16[Nrows,K] @ Wt_bf16[Hout,K]^T + bias)
// Block: 256 threads = 8 waves (wave32). Block tile 128x128, K-step 32.
// Wave grid 4(M) x 2(N); wave tile 32x64 = 2x4 WMMA 16x16 tiles.
// Both tiles staged by TDM, double-buffered on TENSORcnt.
// ---------------------------------------------------------------------------
#define BM 128
#define BN 128
#define BK 32
#define LDT 40   // padded LDS row stride in bf16 elements (80B, 16B aligned)

__global__ __launch_bounds__(256)
void gemm_bias_relu_bf16(const __bf16* __restrict__ A,
                         const __bf16* __restrict__ Wt,
                         const float*  __restrict__ bias,
                         __bf16* __restrict__ C,
                         int Nrows, int K, int Hout, int do_relu) {
    __shared__ __align__(128) __bf16 sA[2 * BM * LDT];
    __shared__ __align__(128) __bf16 sB[2 * BN * LDT];   // sB[n*LDT + k]

    const int t    = threadIdx.x;
    const int lane = t & 31;
    const int wid  = t >> 5;       // 0..7
    const int wm   = wid >> 1;     // 0..3 -> M offset wm*32
    const int wn   = wid & 1;      // 0..1 -> N offset wn*64
    const int l16  = lane & 15;
    const int kh   = lane >> 4;    // K-half select per ISA 16-bit layouts

    const int nb = blockIdx.x * BN;
    const int mb = blockIdx.y * BM;

    v8f acc[2][4] = {};

    const v8i_ g1a = tdm_desc_g1(K, Nrows);   // A tensor [Nrows, K]
    const v8i_ g1b = tdm_desc_g1(K, Hout);    // Wt tensor [Hout, K]
    const unsigned long long gaA =
        (unsigned long long)(const char*)(A + (size_t)mb * K);
    const unsigned long long gaB =
        (unsigned long long)(const char*)(Wt + (size_t)nb * K);
    const unsigned int ldsA = (unsigned int)(size_t)(void*)sA;
    const unsigned int ldsB = (unsigned int)(size_t)(void*)sB;
    const unsigned int bufBytes = BM * LDT * 2;

    const int nIter = K / BK;

    // preload tile pair 0 into buffer 0
    if (t < 32) {
        tdm_load_2d(ldsA, gaA, g1a);
        tdm_load_2d(ldsB, gaB, g1b);
    }

    for (int i = 0; i < nIter; ++i) {
        const int cur = i & 1;
        const int kb  = i * BK;

        if (t < 32) {
            if (i + 1 < nIter) {
                // issue next tile pair into the other buffer (released by the
                // trailing barrier of iteration i-1), then wait for pair i:
                // TDM completes in order per wave -> <=2 pending means the
                // current pair has landed while the next streams in.
                const unsigned int nbuf = (cur ^ 1) * bufBytes;
                tdm_load_2d(ldsA + nbuf, gaA + (unsigned long long)(kb + BK) * 2, g1a);
                tdm_load_2d(ldsB + nbuf, gaB + (unsigned long long)(kb + BK) * 2, g1b);
                __builtin_amdgcn_s_wait_tensorcnt(2);
            } else {
                __builtin_amdgcn_s_wait_tensorcnt(0);
            }
        } else if (i + 2 < nIter) {
            // warm GL2 ahead of the TDM engine (global_prefetch_b8)
            const int kb2 = kb + 2 * BK;
            if (t < 160)
                __builtin_prefetch(A + (size_t)(mb + (t - 32)) * K + kb2, 0, 1);
            else
                __builtin_prefetch(Wt + (size_t)(nb + (t - 160)) * K + kb2, 0, 1);
        }
        __syncthreads();   // tile pair i visible to all waves

        const __bf16* pA = sA + cur * (BM * LDT);
        const __bf16* pB = sB + cur * (BN * LDT);

        // ---- A fragments (ISA 16-bit A 16x32 layout) ----
        // lane: M = l16, K-half = kh. elems 0..7 -> K = kh*8.., 8..15 -> 16+kh*8..
        v16bf afrag[2];
#pragma unroll
        for (int mt = 0; mt < 2; ++mt) {
            const int m = wm * 32 + mt * 16 + l16;
            v8bf lo = *(const v8bf*)&pA[m * LDT + kh * 8];
            v8bf hi = *(const v8bf*)&pA[m * LDT + 16 + kh * 8];
            afrag[mt] = join16(lo, hi);
        }

        // ---- B fragments (per-lane column N = l16; contiguous K in sB) ----
#pragma unroll
        for (int nt = 0; nt < 4; ++nt) {
            const int n = wn * 64 + nt * 16 + l16;
            v8bf blo = *(const v8bf*)&pB[n * LDT + kh * 8];
            v8bf bhi = *(const v8bf*)&pB[n * LDT + 16 + kh * 8];
            v16bf bfrag = join16(blo, bhi);
#pragma unroll
            for (int mt = 0; mt < 2; ++mt) {
                acc[mt][nt] = __builtin_amdgcn_wmma_f32_16x16x32_bf16(
                    /*neg_a=*/false, afrag[mt],
                    /*neg_b=*/false, bfrag,
                    /*c_mod=*/(short)0, acc[mt][nt],
                    /*reuse_a=*/false, /*reuse_b=*/false);
            }
        }
        __syncthreads();   // all waves done reading buffer `cur`
    }

    // ---- epilogue: bias + ReLU + bf16 store ----
    // C/D layout: lanes 0-15 col N=l16 rows r; lanes 16-31 rows 8+r.
#pragma unroll
    for (int mt = 0; mt < 2; ++mt) {
#pragma unroll
        for (int nt = 0; nt < 4; ++nt) {
            const int col = nb + wn * 64 + nt * 16 + l16;
            const float bv = bias[col];
#pragma unroll
            for (int r = 0; r < 8; ++r) {
                const int row = mb + wm * 32 + mt * 16 + kh * 8 + r;
                float v = acc[mt][nt][r] + bv;
                if (do_relu) v = v > 0.0f ? v : 0.0f;
                C[(size_t)row * Hout + col] = (__bf16)v;
            }
        }
    }
}

// ---------------------------------------------------------------------------
// out[row] = sum_k h2[row,k] * W3[k] + b3    (wave-per-row, wave32 shuffle)
// ---------------------------------------------------------------------------
__global__ __launch_bounds__(256)
void gemv_rows(const __bf16* __restrict__ H2, const float* __restrict__ W3,
               const float* __restrict__ b3, float* __restrict__ out, int K) {
    const int lane = threadIdx.x & 31;
    const int wid  = threadIdx.x >> 5;
    const int row  = blockIdx.x * 8 + wid;
    const __bf16* hrow = H2 + (size_t)row * K;

    float acc = 0.0f;
    for (int j = 0; j < K; j += 256) {        // 32 lanes x 8 elems
        const int idx = j + lane * 8;
        uint4 q = *(const uint4*)(hrow + idx);
        const __bf16* p = (const __bf16*)&q;
#pragma unroll
        for (int i = 0; i < 8; ++i) acc += (float)p[i] * W3[idx + i];
    }
#pragma unroll
    for (int o = 16; o > 0; o >>= 1) acc += __shfl_down(acc, o, 32);
    if (lane == 0) out[row] = acc + b3[0];
}

// ---------------------------------------------------------------------------
// segment reduction + final combine
// ---------------------------------------------------------------------------
__global__ void zero_f32(float* __restrict__ p, int n) {
    int i = blockIdx.x * blockDim.x + threadIdx.x;
    if (i < n) p[i] = 0.0f;
}

__global__ void seg_accum(const float* __restrict__ of, const float* __restrict__ ow,
                          const int* __restrict__ seg,
                          float* __restrict__ so, float* __restrict__ sw, int n) {
    int i = blockIdx.x * blockDim.x + threadIdx.x;
    if (i < n) {
        int s = seg[i];
        atomicAdd(&so[s], of[i]);
        atomicAdd(&sw[s], ow[i]);
    }
}

__global__ void finalize(const float* __restrict__ of, const float* __restrict__ ow,
                         const int* __restrict__ seg,
                         const float* __restrict__ so, const float* __restrict__ sw,
                         float* __restrict__ out, int n) {
    int i = blockIdx.x * blockDim.x + threadIdx.x;
    if (i < n) {
        int s = seg[i];
        float ratio = so[s] / sw[s];
        out[i] = of[i] - ow[i] * ratio;
    }
}

// ---------------------------------------------------------------------------
// host launcher
// ---------------------------------------------------------------------------
extern "C" void kernel_launch(void* const* d_in, const int* in_sizes, int n_in,
                              void* d_out, int out_size, void* d_ws, size_t ws_size,
                              hipStream_t stream) {
    constexpr int N = 65536, D = 1024, H = 2048, M = 4096;

    const float* hidden = (const float*)d_in[0];
    const float* fW1 = (const float*)d_in[1];  const float* fb1 = (const float*)d_in[2];
    const float* fW2 = (const float*)d_in[3];  const float* fb2 = (const float*)d_in[4];
    const float* fW3 = (const float*)d_in[5];  const float* fb3 = (const float*)d_in[6];
    const float* rW1 = (const float*)d_in[7];  const float* rb1 = (const float*)d_in[8];
    const float* rW2 = (const float*)d_in[9];  const float* rb2 = (const float*)d_in[10];
    const float* rW3 = (const float*)d_in[11]; const float* rb3 = (const float*)d_in[12];
    const int*   seg = (const int*)d_in[13];
    float* out = (float*)d_out;

    // workspace carve-out (256B aligned regions)
    char* base = (char*)d_ws;
    size_t off = 0;
    auto alloc = [&](size_t bytes) -> char* {
        char* p = base + off;
        off = (off + bytes + 255) & ~(size_t)255;
        return p;
    };
    __bf16* hb   = (__bf16*)alloc((size_t)N * D * 2);   // hidden bf16
    __bf16* w1ft = (__bf16*)alloc((size_t)D * H * 2);   // transposed [H,D]
    __bf16* w2ft = (__bf16*)alloc((size_t)H * H * 2);   // transposed [H,H]
    __bf16* w1rt = (__bf16*)alloc((size_t)D * H * 2);
    __bf16* w2rt = (__bf16*)alloc((size_t)H * H * 2);
    __bf16* h1   = (__bf16*)alloc((size_t)N * H * 2);   // reused by both paths
    __bf16* h2   = (__bf16*)alloc((size_t)N * H * 2);   // reused by both paths
    float*  of   = (float*)alloc((size_t)N * 4);
    float*  ow   = (float*)alloc((size_t)N * 4);
    float*  so   = (float*)alloc((size_t)M * 4);
    float*  sw   = (float*)alloc((size_t)M * 4);
    (void)ws_size; (void)in_sizes; (void)n_in; (void)out_size;

    // 1) conversions (weights transposed so GEMM B-tiles are TDM-friendly)
    cvt_f32_bf16<<<4096, 256, 0, stream>>>(hidden, hb, (size_t)N * D);
    dim3 t1(H / 32, D / 32), t2(H / 32, H / 32);
    cvt_transpose_f32_bf16<<<t1, 256, 0, stream>>>(fW1, w1ft, D, H);
    cvt_transpose_f32_bf16<<<t2, 256, 0, stream>>>(fW2, w2ft, H, H);
    cvt_transpose_f32_bf16<<<t1, 256, 0, stream>>>(rW1, w1rt, D, H);
    cvt_transpose_f32_bf16<<<t2, 256, 0, stream>>>(rW2, w2rt, H, H);

    dim3 gemm_grid(H / BN, N / BM);   // (16, 512)

    // 2-4) ffn path
    gemm_bias_relu_bf16<<<gemm_grid, 256, 0, stream>>>(hb, w1ft, fb1, h1, N, D, H, 1);
    gemm_bias_relu_bf16<<<gemm_grid, 256, 0, stream>>>(h1, w2ft, fb2, h2, N, H, H, 1);
    gemv_rows<<<N / 8, 256, 0, stream>>>(h2, fW3, fb3, of, H);

    // 2-4) weight-regression path (reuse h1/h2)
    gemm_bias_relu_bf16<<<gemm_grid, 256, 0, stream>>>(hb, w1rt, rb1, h1, N, D, H, 1);
    gemm_bias_relu_bf16<<<gemm_grid, 256, 0, stream>>>(h1, w2rt, rb2, h2, N, H, H, 1);
    gemv_rows<<<N / 8, 256, 0, stream>>>(h2, rW3, rb3, ow, H);

    // 5) segment sums + final combine
    zero_f32<<<M / 256, 256, 0, stream>>>(so, M);
    zero_f32<<<M / 256, 256, 0, stream>>>(sw, M);
    seg_accum<<<N / 256, 256, 0, stream>>>(of, ow, seg, so, sw, N);
    finalize<<<N / 256, 256, 0, stream>>>(of, ow, seg, so, sw, out, N);
}
